// Attention_14559939133603
// MI455X (gfx1250) — matrix operand order repaired and checked by reading, not compiled, not run on hardware
//
#include <hip/hip_runtime.h>
#include <hip/hip_bf16.h>

typedef __attribute__((ext_vector_type(16))) _Float16 v16h;
typedef __attribute__((ext_vector_type(8)))  _Float16 v8h;
typedef __attribute__((ext_vector_type(8)))  float    v8f;

#define SEQ_N 4096
#define CDIM  512
#define NHEAD 8
#define HDIM  64
#define BATCH 2
#define MROWS (BATCH * SEQ_N)   // 8192

// ---------------- WMMA helpers (A/B frags in contiguous k order: k = 16*(lane/16) + s) ----------------

__device__ __forceinline__ v8f wmma_f16(v16h a, v16h b, v8f c) {
    return __builtin_amdgcn_wmma_f32_16x16x32_f16(
        /*neg_a=*/false, a, /*neg_b=*/false, b,
        /*c_mod=*/(short)0, c, /*reuse_a=*/false, /*reuse_b=*/false);
}

// A-matrix fragment 16x32 (M x K) from row-major memory (K contiguous, row
// stride ld). lane<16: row=lane, K = 0..15 contiguous;
// lane>=16: same row, K = 16..31.
__device__ __forceinline__ v16h load_fragA(const _Float16* __restrict__ p, int ld) {
    int lane = threadIdx.x & 31;
    const _Float16* q = p + (size_t)(lane & 15) * ld + ((lane >> 4) << 4);
    v8h lo = *(const v8h*)(q);        // K: kofs .. kofs+7
    v8h hi = *(const v8h*)(q + 8);    // K: kofs+8 .. kofs+15
    v16h r;
#pragma unroll
    for (int i = 0; i < 8; ++i) { r[i] = lo[i]; r[i + 8] = hi[i]; }
    return r;
}

// B-matrix fragment 32x16 (K x N) from B^T row-major memory: mem[n][k], k
// contiguous, row stride ld. lane<16: col N=lane, K=0..15 contiguous;
// lane>=16: col N=lane-16, K=16..31.
__device__ __forceinline__ v16h load_fragB(const _Float16* __restrict__ p, int ld) {
    int lane = threadIdx.x & 31;
    const _Float16* q = p + (size_t)(lane & 15) * ld + ((lane >> 4) << 4);
    return *(const v16h*)q;           // 32B contiguous
}

// width-16 row reductions (C/D layout: lanes 0-15 and 16-31 hold distinct rows)
__device__ __forceinline__ float rowmax16(float v) {
#pragma unroll
    for (int off = 8; off >= 1; off >>= 1) v = fmaxf(v, __shfl_xor(v, off, 16));
    return v;
}
__device__ __forceinline__ float rowsum16(float v) {
#pragma unroll
    for (int off = 8; off >= 1; off >>= 1) v += __shfl_xor(v, off, 16);
    return v;
}

// ---------------- Stage 0: fp32 -> f16 ----------------

__global__ void cvt_f32_to_f16(const float* __restrict__ in,
                               _Float16* __restrict__ out, int n) {
    int i = blockIdx.x * blockDim.x + threadIdx.x;
    if (i < n) out[i] = (_Float16)in[i];
}

// ---------------- Stage 1: fused QKV projection ----------------
// y[m,n] = sum_k x[m,k] * W[n,k]; Q scaled by 0.125; V stored transposed
// as vt[(b*8+h)*64+d][m%4096].
// Wave tile: 32(M) x 64(N), software-pipelined K loop.
// grid = (32, 8, 3), block = 256 (8 waves).

__global__ __launch_bounds__(256) void qkv_gemm(
    const _Float16* __restrict__ xh,
    const _Float16* __restrict__ wq, const _Float16* __restrict__ wk,
    const _Float16* __restrict__ wv,
    _Float16* __restrict__ qb, _Float16* __restrict__ kb,
    _Float16* __restrict__ vt) {
    int wave = threadIdx.x >> 5;
    int m0 = blockIdx.x * 256 + wave * 32;
    int n0 = blockIdx.y * 64;
    int which = blockIdx.z;
    const _Float16* w = (which == 0) ? wq : (which == 1) ? wk : wv;
    const _Float16* xp = xh + (size_t)m0 * CDIM;
    const _Float16* wp = w + (size_t)n0 * CDIM;

    v8f acc[8] = {};
    // prologue: fragments for k=0
    v16h a0 = load_fragA(xp, CDIM);
    v16h a1 = load_fragA(xp + (size_t)16 * CDIM, CDIM);
    v16h b0 = load_fragB(wp, CDIM);
    v16h b1 = load_fragB(wp + (size_t)16 * CDIM, CDIM);
    v16h b2 = load_fragB(wp + (size_t)32 * CDIM, CDIM);
    v16h b3 = load_fragB(wp + (size_t)48 * CDIM, CDIM);

    for (int kk = 32; kk < CDIM; kk += 32) {
        // issue next-k loads before consuming current fragments
        v16h na0 = load_fragA(xp + kk, CDIM);
        v16h na1 = load_fragA(xp + (size_t)16 * CDIM + kk, CDIM);
        v16h nb0 = load_fragB(wp + kk, CDIM);
        v16h nb1 = load_fragB(wp + (size_t)16 * CDIM + kk, CDIM);
        v16h nb2 = load_fragB(wp + (size_t)32 * CDIM + kk, CDIM);
        v16h nb3 = load_fragB(wp + (size_t)48 * CDIM + kk, CDIM);
        acc[0] = wmma_f16(a0, b0, acc[0]);
        acc[1] = wmma_f16(a0, b1, acc[1]);
        acc[2] = wmma_f16(a0, b2, acc[2]);
        acc[3] = wmma_f16(a0, b3, acc[3]);
        acc[4] = wmma_f16(a1, b0, acc[4]);
        acc[5] = wmma_f16(a1, b1, acc[5]);
        acc[6] = wmma_f16(a1, b2, acc[6]);
        acc[7] = wmma_f16(a1, b3, acc[7]);
        a0 = na0; a1 = na1; b0 = nb0; b1 = nb1; b2 = nb2; b3 = nb3;
    }
    acc[0] = wmma_f16(a0, b0, acc[0]);
    acc[1] = wmma_f16(a0, b1, acc[1]);
    acc[2] = wmma_f16(a0, b2, acc[2]);
    acc[3] = wmma_f16(a0, b3, acc[3]);
    acc[4] = wmma_f16(a1, b0, acc[4]);
    acc[5] = wmma_f16(a1, b1, acc[5]);
    acc[6] = wmma_f16(a1, b2, acc[6]);
    acc[7] = wmma_f16(a1, b3, acc[7]);

    int lane = threadIdx.x & 31;
    int col  = lane & 15;
    int rofs = (lane >> 4) ? 8 : 0;
    if (which < 2) {
        _Float16* dst = (which == 0) ? qb : kb;
        float scale = (which == 0) ? 0.125f : 1.0f;  // fold SCALE into Q
#pragma unroll
        for (int mt = 0; mt < 2; ++mt)
#pragma unroll
            for (int t = 0; t < 4; ++t)
#pragma unroll
                for (int r = 0; r < 8; ++r)
                    dst[(size_t)(m0 + 16 * mt + r + rofs) * CDIM + n0 + 16 * t + col] =
                        (_Float16)(acc[mt * 4 + t][r] * scale);
    } else {
#pragma unroll
        for (int mt = 0; mt < 2; ++mt) {
            int m = m0 + 16 * mt + rofs;   // 8 consecutive rows from here
            int bidx = m >> 12;            // /4096
            int mm = m & (SEQ_N - 1);
#pragma unroll
            for (int t = 0; t < 4; ++t) {
                int n = n0 + 16 * t + col;  // 0..511
                int h = n >> 6, d = n & 63;
                v8h pk;
#pragma unroll
                for (int r = 0; r < 8; ++r) pk[r] = (_Float16)acc[mt * 4 + t][r];
                *(v8h*)(vt + ((size_t)((bidx * NHEAD + h) * HDIM + d)) * SEQ_N + mm) = pk;
            }
        }
    }
}

// ---------------- Stage 2: flash attention ----------------
// grid = (32, 8, 2) -> (128 query rows, head, batch), block = 256 (8 waves).
// Each wave: 16 query rows, streams 32-key blocks with online softmax.
// Pipelining: V frags for block j issued at top of iteration (consumed after
// softmax); K frags for block j+32 prefetched right after the S WMMAs.

__global__ __launch_bounds__(256) void flash_attn(
    const _Float16* __restrict__ qb, const _Float16* __restrict__ kbuf,
    const _Float16* __restrict__ vt, _Float16* __restrict__ ao) {
    __shared__ _Float16 ldsP[8][16][32];   // per-wave P tile (layout bridge)

    int wave = threadIdx.x >> 5;
    int lane = threadIdx.x & 31;
    int h = blockIdx.y, b = blockIdx.z;
    int q0 = blockIdx.x * 128 + wave * 16;

    const _Float16* qbase = qb   + (size_t)(b * SEQ_N) * CDIM + h * HDIM;
    const _Float16* kbase = kbuf + (size_t)(b * SEQ_N) * CDIM + h * HDIM;
    const _Float16* vbase = vt   + (size_t)((b * NHEAD + h) * HDIM) * SEQ_N;

    // Q fragments for this wave's 16 rows (D=64 -> two K-steps of 32)
    v16h qa0 = load_fragA(qbase + (size_t)q0 * CDIM,      CDIM);
    v16h qa1 = load_fragA(qbase + (size_t)q0 * CDIM + 32, CDIM);

    v8f o[4] = {};
    float mrow[8], lrow[8];
#pragma unroll
    for (int r = 0; r < 8; ++r) { mrow[r] = -1e30f; lrow[r] = 0.0f; }

    _Float16* pl = &ldsP[wave][0][0];
    int col  = lane & 15;
    int rofs = (lane >> 4) ? 8 : 0;

    // prologue: K fragments for block j=0
    // kb0/kb1: key rows j..j+15 (d 0..31 / 32..63); kb2/kb3: rows j+16..j+31
    v16h kb0 = load_fragB(kbase, CDIM);
    v16h kb1 = load_fragB(kbase + 32, CDIM);
    v16h kb2 = load_fragB(kbase + (size_t)16 * CDIM, CDIM);
    v16h kb3 = load_fragB(kbase + (size_t)16 * CDIM + 32, CDIM);

    for (int j = 0; j < SEQ_N; j += 32) {
        // V fragments for the CURRENT block, issued early (used after softmax)
        v16h vb0 = load_fragB(vbase + j,                        SEQ_N);
        v16h vb1 = load_fragB(vbase + (size_t)16 * SEQ_N + j,   SEQ_N);
        v16h vb2 = load_fragB(vbase + (size_t)32 * SEQ_N + j,   SEQ_N);
        v16h vb3 = load_fragB(vbase + (size_t)48 * SEQ_N + j,   SEQ_N);

        // S = Q K^T for 16 rows x 32 keys (SCALE already folded into Q)
        v8f s0 = {}, s1 = {};
        s0 = wmma_f16(qa0, kb0, s0); s0 = wmma_f16(qa1, kb1, s0);
        s1 = wmma_f16(qa0, kb2, s1); s1 = wmma_f16(qa1, kb3, s1);

        // prefetch K fragments for the NEXT block. On the final iteration this
        // reads ~32KB past the K buffer, which falls inside the adjacent Vt
        // workspace region (allocated; values never consumed).
        int jn = j + 32;
        v16h nk0 = load_fragB(kbase + (size_t)jn * CDIM, CDIM);
        v16h nk1 = load_fragB(kbase + (size_t)jn * CDIM + 32, CDIM);
        v16h nk2 = load_fragB(kbase + (size_t)(jn + 16) * CDIM, CDIM);
        v16h nk3 = load_fragB(kbase + (size_t)(jn + 16) * CDIM + 32, CDIM);

        // online softmax per row (rows live on vgpr index; lanes = key cols)
#pragma unroll
        for (int r = 0; r < 8; ++r) {
            float mij  = rowmax16(fmaxf(s0[r], s1[r]));
            float mnew = fmaxf(mrow[r], mij);
            float alpha = __expf(mrow[r] - mnew);
            float p0 = __expf(s0[r] - mnew);
            float p1 = __expf(s1[r] - mnew);
            lrow[r] = lrow[r] * alpha + rowsum16(p0 + p1);
            mrow[r] = mnew;
#pragma unroll
            for (int t = 0; t < 4; ++t) o[t][r] *= alpha;
            // C/D layout -> row-major P tile in LDS
            pl[(r + rofs) * 32 + col]      = (_Float16)p0;
            pl[(r + rofs) * 32 + col + 16] = (_Float16)p1;
        }
        // O += P * V  (Vt is [d][m]: B-frag loads are contiguous)
        v16h pa = load_fragA(pl, 32);
        o[0] = wmma_f16(pa, vb0, o[0]);
        o[1] = wmma_f16(pa, vb1, o[1]);
        o[2] = wmma_f16(pa, vb2, o[2]);
        o[3] = wmma_f16(pa, vb3, o[3]);

        kb0 = nk0; kb1 = nk1; kb2 = nk2; kb3 = nk3;
    }

    // normalize and store attention output (f16, head-interleaved [B,N,C])
    _Float16* aob = ao + (size_t)(b * SEQ_N + q0) * CDIM + h * HDIM;
#pragma unroll
    for (int r = 0; r < 8; ++r) {
        float inv = 1.0f / lrow[r];
#pragma unroll
        for (int t = 0; t < 4; ++t)
            aob[(size_t)(r + rofs) * CDIM + 16 * t + col] = (_Float16)(o[t][r] * inv);
    }
}

// ---------------- Stage 3: output projection + bias (fp32 out) ----------------
// Wave tile 32x64, pipelined like qkv_gemm. grid = (32, 8), block = 256.

__global__ __launch_bounds__(256) void out_proj(
    const _Float16* __restrict__ ao, const _Float16* __restrict__ woh,
    const float* __restrict__ bo, float* __restrict__ out) {
    int wave = threadIdx.x >> 5;
    int m0 = blockIdx.x * 256 + wave * 32;
    int n0 = blockIdx.y * 64;
    const _Float16* xp = ao + (size_t)m0 * CDIM;
    const _Float16* wp = woh + (size_t)n0 * CDIM;

    v8f acc[8] = {};
    v16h a0 = load_fragA(xp, CDIM);
    v16h a1 = load_fragA(xp + (size_t)16 * CDIM, CDIM);
    v16h b0 = load_fragB(wp, CDIM);
    v16h b1 = load_fragB(wp + (size_t)16 * CDIM, CDIM);
    v16h b2 = load_fragB(wp + (size_t)32 * CDIM, CDIM);
    v16h b3 = load_fragB(wp + (size_t)48 * CDIM, CDIM);

    for (int kk = 32; kk < CDIM; kk += 32) {
        v16h na0 = load_fragA(xp + kk, CDIM);
        v16h na1 = load_fragA(xp + (size_t)16 * CDIM + kk, CDIM);
        v16h nb0 = load_fragB(wp + kk, CDIM);
        v16h nb1 = load_fragB(wp + (size_t)16 * CDIM + kk, CDIM);
        v16h nb2 = load_fragB(wp + (size_t)32 * CDIM + kk, CDIM);
        v16h nb3 = load_fragB(wp + (size_t)48 * CDIM + kk, CDIM);
        acc[0] = wmma_f16(a0, b0, acc[0]);
        acc[1] = wmma_f16(a0, b1, acc[1]);
        acc[2] = wmma_f16(a0, b2, acc[2]);
        acc[3] = wmma_f16(a0, b3, acc[3]);
        acc[4] = wmma_f16(a1, b0, acc[4]);
        acc[5] = wmma_f16(a1, b1, acc[5]);
        acc[6] = wmma_f16(a1, b2, acc[6]);
        acc[7] = wmma_f16(a1, b3, acc[7]);
        a0 = na0; a1 = na1; b0 = nb0; b1 = nb1; b2 = nb2; b3 = nb3;
    }
    acc[0] = wmma_f16(a0, b0, acc[0]);
    acc[1] = wmma_f16(a0, b1, acc[1]);
    acc[2] = wmma_f16(a0, b2, acc[2]);
    acc[3] = wmma_f16(a0, b3, acc[3]);
    acc[4] = wmma_f16(a1, b0, acc[4]);
    acc[5] = wmma_f16(a1, b1, acc[5]);
    acc[6] = wmma_f16(a1, b2, acc[6]);
    acc[7] = wmma_f16(a1, b3, acc[7]);

    int lane = threadIdx.x & 31;
    int col  = lane & 15;
    int rofs = (lane >> 4) ? 8 : 0;
#pragma unroll
    for (int mt = 0; mt < 2; ++mt)
#pragma unroll
        for (int t = 0; t < 4; ++t) {
            float bias = bo[n0 + 16 * t + col];
#pragma unroll
            for (int r = 0; r < 8; ++r)
                out[(size_t)(m0 + 16 * mt + r + rofs) * CDIM + n0 + 16 * t + col] =
                    acc[mt * 4 + t][r] + bias;
        }
}

// ---------------- launcher ----------------

extern "C" void kernel_launch(void* const* d_in, const int* in_sizes, int n_in,
                              void* d_out, int out_size, void* d_ws, size_t ws_size,
                              hipStream_t stream) {
    (void)in_sizes; (void)n_in; (void)out_size; (void)ws_size;
    const float* x  = (const float*)d_in[0];
    const float* Wq = (const float*)d_in[1];
    const float* Wk = (const float*)d_in[2];
    const float* Wv = (const float*)d_in[3];
    const float* Wo = (const float*)d_in[4];
    const float* bo = (const float*)d_in[5];

    const size_t NX = (size_t)MROWS * CDIM;   // 4,194,304
    const size_t NW = (size_t)CDIM * CDIM;    //   262,144

    _Float16* ws  = (_Float16*)d_ws;
    _Float16* xh  = ws;                 // [8192,512]
    _Float16* wqh = xh  + NX;           // [512,512] (n-major, k contiguous)
    _Float16* wkh = wqh + NW;
    _Float16* wvh = wkh + NW;
    _Float16* woh = wvh + NW;
    _Float16* qb  = woh + NW;           // [B*N,512] f16 (scaled by 0.125)
    _Float16* kb  = qb  + NX;           // [B*N,512] f16
    _Float16* vt  = kb  + NX;           // [B,H,D,N] f16 (transposed V)
    _Float16* aob = vt  + NX;           // [B*N,512] f16 attention output

    // Stage 0: precision conversion
    cvt_f32_to_f16<<<dim3((NX + 255) / 256), 256, 0, stream>>>(x,  xh,  (int)NX);
    cvt_f32_to_f16<<<dim3((NW + 255) / 256), 256, 0, stream>>>(Wq, wqh, (int)NW);
    cvt_f32_to_f16<<<dim3((NW + 255) / 256), 256, 0, stream>>>(Wk, wkh, (int)NW);
    cvt_f32_to_f16<<<dim3((NW + 255) / 256), 256, 0, stream>>>(Wv, wvh, (int)NW);
    cvt_f32_to_f16<<<dim3((NW + 255) / 256), 256, 0, stream>>>(Wo, woh, (int)NW);

    // Stage 1: fused QKV projection (z selects Q/K/V)
    qkv_gemm<<<dim3(MROWS / 256, CDIM / 64, 3), 256, 0, stream>>>(
        xh, wqh, wkh, wvh, qb, kb, vt);

    // Stage 2: flash attention
    flash_attn<<<dim3(SEQ_N / 128, NHEAD, BATCH), 256, 0, stream>>>(
        qb, kb, vt, aob);

    // Stage 3: output projection + bias
    out_proj<<<dim3(MROWS / 256, CDIM / 64, 1), 256, 0, stream>>>(
        aob, woh, bo, (float*)d_out);
}
